// SymmetricFullCoupling_33071248180131
// MI455X (gfx1250) — matrix operand order repaired and checked by reading, not compiled, not run on hardware
//
#include <hip/hip_runtime.h>

// ---------------------------------------------------------------------------
// Types for CDNA5 WMMA (wave32): v16bf A/B fragments, v8f accumulator.
// ---------------------------------------------------------------------------
typedef __attribute__((ext_vector_type(16))) __bf16       v16bf;
typedef __attribute__((ext_vector_type(8)))  float        v8f;
typedef __attribute__((ext_vector_type(4)))  unsigned int v4u;

union FragBF {
  v16bf v;
  v4u   u[2];
};

// Problem sizes (compile-time constants from the reference).
#define NB   8192   // batch
#define ND   1024   // D1 == D2
#define NH   4096   // hidden

// GEMM tiling: 256 threads = 8 wave32s; block tile 128x128, K-tile 32.
#define BM  128
#define BN  128
#define BK  32
#define LDT 40      // padded LDS row stride (elements) -> 80B rows, 16B aligned

__device__ __forceinline__ unsigned short f2bf(float f) {
  // round-to-nearest-even fp32 -> bf16 (bit pattern as ushort)
  unsigned int u = __float_as_uint(f);
  u += 0x7fffu + ((u >> 16) & 1u);
  return (unsigned short)(u >> 16);
}

// Low 32 bits of a generic pointer to LDS == byte offset within the
// workgroup's LDS allocation (amdgcn addrspacecast construction).
__device__ __forceinline__ unsigned lds_off(const void* p) {
  return (unsigned)(unsigned long long)p;
}

// gfx1250 async global->LDS copy: 16 bytes per lane, no VGPR round-trip.
// Tracked by ASYNCcnt (in-order retire) -> software pipelining via
// s_wait_asynccnt with a nonzero threshold.
__device__ __forceinline__ void async_cp16(unsigned lds_byte, const void* g) {
  asm volatile("global_load_async_to_lds_b128 %0, %1, off"
               :: "v"(lds_byte), "v"(g)
               : "memory");
}
__device__ __forceinline__ void wait_async_le4() {
  asm volatile("s_wait_asynccnt 0x4" ::: "memory");
}
__device__ __forceinline__ void wait_async_0() {
  asm volatile("s_wait_asynccnt 0x0" ::: "memory");
}

// ---------------------------------------------------------------------------
// C[M,N] = act(A[M,K](bf16) @ Bt[N,K](bf16)^T + bias[N])
// Both operands K-major bf16 -> both LDS tiles staged with pure async b128
// copies, double-buffered. Each wave computes a 32(M) x 64(N) sub-tile
// = 2x4 WMMA tiles (8 x v_wmma_f32_16x16x32_bf16 per K-step).
// ---------------------------------------------------------------------------
template <bool RELU, bool OUT_BF16>
__global__ __launch_bounds__(256)
void gemm_bf16_wmma(const unsigned short* __restrict__ A,
                    const unsigned short* __restrict__ Bt,
                    const float* __restrict__ bias,
                    void* __restrict__ Cout,
                    int N, int K)
{
  __shared__ __align__(16) unsigned short As[2][BM * LDT];
  __shared__ __align__(16) unsigned short Bs[2][BN * LDT];   // [n][k]

  const int tid  = threadIdx.x;
  const int lane = tid & 31;
  const int wave = tid >> 5;
  const int wm   = (wave & 3) * 32;   // wave M offset in block tile
  const int wn   = (wave >> 2) * 64;  // wave N offset in block tile
  const int half = lane >> 4;         // 0: lanes 0-15, 1: lanes 16-31
  const int l16  = lane & 15;

  const int blockM = blockIdx.y * BM;
  const int blockN = blockIdx.x * BN;

  // Per-thread staging slots: tile = 128 rows x 64B = 512 x 16B chunks,
  // 256 threads -> 2 chunks each per operand (4 async issues per tile).
  auto issue_tile = [&](int k0, int buf) {
#pragma unroll
    for (int j = 0; j < 2; ++j) {
      int i   = tid + 256 * j;
      int row = i >> 2;
      int c8  = (i & 3) * 8;
      async_cp16(lds_off(&As[buf][row * LDT + c8]),
                 A  + (size_t)(blockM + row) * K + k0 + c8);
      async_cp16(lds_off(&Bs[buf][row * LDT + c8]),
                 Bt + (size_t)(blockN + row) * K + k0 + c8);
    }
  };

  v8f acc[2][4];
#pragma unroll
  for (int mt = 0; mt < 2; ++mt)
#pragma unroll
    for (int nt = 0; nt < 4; ++nt)
      acc[mt][nt] = (v8f){0.f, 0.f, 0.f, 0.f, 0.f, 0.f, 0.f, 0.f};

  issue_tile(0, 0);
  const int nIter = K / BK;
  for (int it = 0; it < nIter; ++it) {
    const int cur = it & 1;
    if (it + 1 < nIter) {
      issue_tile((it + 1) * BK, cur ^ 1);  // prefetch next tile (async)
      wait_async_le4();                    // tile `it` has fully landed
    } else {
      wait_async_0();
    }
    __syncthreads();

    // ---- fragments per ISA 16-bit layouts ----
    FragBF a[2], b[4];
#pragma unroll
    for (int mt = 0; mt < 2; ++mt) {
      // A 16x32: lanes 0-15 hold K{0..7,16..23}, lanes 16-31 K{8..15,24..31}
      const unsigned short* p = &As[cur][(wm + mt * 16 + l16) * LDT + half * 8];
      a[mt].u[0] = *reinterpret_cast<const v4u*>(p);
      a[mt].u[1] = *reinterpret_cast<const v4u*>(p + 16);
    }
#pragma unroll
    for (int nt = 0; nt < 4; ++nt) {
      // B 32x16: lanes 0-15 hold K{0..7,16..23}, lanes 16-31 K{8..15,24..31}
      const unsigned short* p = &Bs[cur][(wn + nt * 16 + l16) * LDT + half * 8];
      b[nt].u[0] = *reinterpret_cast<const v4u*>(p);
      b[nt].u[1] = *reinterpret_cast<const v4u*>(p + 16);
    }

#pragma unroll
    for (int mt = 0; mt < 2; ++mt)
#pragma unroll
      for (int nt = 0; nt < 4; ++nt)
        acc[mt][nt] = __builtin_amdgcn_wmma_f32_16x16x32_bf16(
            false, a[mt].v, false, b[nt].v, (short)0, acc[mt][nt], false, false);
    __syncthreads();   // all waves done reading buf before it is re-filled
  }

  // ---- epilogue: bias (+ReLU), store fp32 or bf16 ----
  // C layout: VGPR v -> row = half*8 + v, col = l16 (16x16 f32 C layout)
#pragma unroll
  for (int nt = 0; nt < 4; ++nt) {
    int   col = blockN + wn + nt * 16 + l16;
    float bv  = bias[col];
#pragma unroll
    for (int mt = 0; mt < 2; ++mt) {
#pragma unroll
      for (int v = 0; v < 8; ++v) {
        int   row = blockM + wm + mt * 16 + half * 8 + v;
        float val = acc[mt][nt][v] + bv;
        if (RELU) val = fmaxf(val, 0.f);
        if (OUT_BF16)
          reinterpret_cast<unsigned short*>(Cout)[(size_t)row * N + col] = f2bf(val);
        else
          reinterpret_cast<float*>(Cout)[(size_t)row * N + col] = val;
      }
    }
  }
}

// ---------------------------------------------------------------------------
// Weight prep: W fp32 [K x N] -> Wt bf16 [N x K] (transpose + convert),
// done once per call so the GEMM hot loop has zero conversion VALU.
// 32x32 LDS-tiled transpose, 256 threads.
// ---------------------------------------------------------------------------
__global__ __launch_bounds__(256)
void transpose_cvt_kernel(const float* __restrict__ W,
                          unsigned short* __restrict__ Wt,
                          int K, int N)
{
  __shared__ float tile[32][33];
  const int kb = blockIdx.y * 32;
  const int nb = blockIdx.x * 32;
  const int tx = threadIdx.x & 31;
  const int ty = threadIdx.x >> 5;     // 8 rows per pass
#pragma unroll
  for (int r = 0; r < 32; r += 8)
    tile[ty + r][tx] = W[(size_t)(kb + ty + r) * N + nb + tx];
  __syncthreads();
#pragma unroll
  for (int r = 0; r < 32; r += 8)
    Wt[(size_t)(nb + ty + r) * K + kb + tx] = f2bf(tile[tx][ty + r]);
}

// ---------------------------------------------------------------------------
// x2 (columns 1024..2047 of x) -> bf16 workspace
// ---------------------------------------------------------------------------
__global__ __launch_bounds__(256)
void cvt_x2_kernel(const float* __restrict__ x, unsigned short* __restrict__ x2)
{
  size_t i   = (size_t)blockIdx.x * blockDim.x + threadIdx.x;  // over NB*ND
  size_t row = i >> 10;
  size_t col = i & 1023;
  x2[i] = f2bf(x[row * (size_t)(2 * ND) + ND + col]);
}

// ---------------------------------------------------------------------------
// Coupling: s = tanh(s_pre); y = x_half * exp(s) + t; log_det row-sum of s.
// One block per row. Phase 1 (accumulate=0): also emit y1 as bf16 for the
// next GEMMs and *store* log_det (clears poison). Phase 2: accumulate.
// ---------------------------------------------------------------------------
__global__ __launch_bounds__(256)
void coupling_kernel(const float* __restrict__ x,
                     const float* __restrict__ s_pre,
                     const float* __restrict__ t_val,
                     float* __restrict__ y_out,          // NB x 2D (d_out)
                     unsigned short* __restrict__ y_bf,  // NB x D or nullptr
                     float* __restrict__ log_det,
                     int col_off, int accumulate)
{
  __shared__ float red[256];
  const int row = blockIdx.x;
  float lsum = 0.f;
  for (int c = threadIdx.x; c < ND; c += 256) {
    float pre = s_pre[(size_t)row * ND + c];
    float s   = tanhf(pre);                     // SCALE_LIMIT == 1.0
    float t   = t_val[(size_t)row * ND + c];
    float xin = x[(size_t)row * (2 * ND) + col_off + c];
    float y   = xin * expf(s) + t;
    y_out[(size_t)row * (2 * ND) + col_off + c] = y;
    if (y_bf) y_bf[(size_t)row * ND + c] = f2bf(y);
    lsum += s;
  }
  red[threadIdx.x] = lsum;
  __syncthreads();
  for (int off = 128; off > 0; off >>= 1) {
    if ((int)threadIdx.x < off) red[threadIdx.x] += red[threadIdx.x + off];
    __syncthreads();
  }
  if (threadIdx.x == 0) {
    if (accumulate) log_det[row] += red[0];
    else            log_det[row]  = red[0];
  }
}

// ---------------------------------------------------------------------------
// Launch: 8 weight transposes + cvt -> 4 GEMMs -> coupling1 -> 4 GEMMs ->
// coupling2, all ordered on one stream.
// ---------------------------------------------------------------------------
extern "C" void kernel_launch(void* const* d_in, const int* in_sizes, int n_in,
                              void* d_out, int out_size, void* d_ws, size_t ws_size,
                              hipStream_t stream)
{
  (void)in_sizes; (void)n_in; (void)out_size; (void)ws_size;

  const float* x     = (const float*)d_in[0];
  const float* s1_W1 = (const float*)d_in[1];
  const float* s1_b1 = (const float*)d_in[2];
  const float* s1_W2 = (const float*)d_in[3];
  const float* s1_b2 = (const float*)d_in[4];
  const float* t1_W1 = (const float*)d_in[5];
  const float* t1_b1 = (const float*)d_in[6];
  const float* t1_W2 = (const float*)d_in[7];
  const float* t1_b2 = (const float*)d_in[8];
  const float* s2_W1 = (const float*)d_in[9];
  const float* s2_b1 = (const float*)d_in[10];
  const float* s2_W2 = (const float*)d_in[11];
  const float* s2_b2 = (const float*)d_in[12];
  const float* t2_W1 = (const float*)d_in[13];
  const float* t2_b1 = (const float*)d_in[14];
  const float* t2_W2 = (const float*)d_in[15];
  const float* t2_b2 = (const float*)d_in[16];

  // Workspace carve-up: x2 16MB | h_s 64MB | h_t 64MB | s_pre 32MB |
  // t_val 32MB | y1 16MB | 8 x Wt 8MB = 64MB  == 288MB total.
  char* ws = (char*)d_ws;
  unsigned short* x2_bf = (unsigned short*)ws; ws += (size_t)NB * ND * 2;
  unsigned short* h_s   = (unsigned short*)ws; ws += (size_t)NB * NH * 2;
  unsigned short* h_t   = (unsigned short*)ws; ws += (size_t)NB * NH * 2;
  float*          s_pre = (float*)ws;          ws += (size_t)NB * ND * 4;
  float*          t_val = (float*)ws;          ws += (size_t)NB * ND * 4;
  unsigned short* y1_bf = (unsigned short*)ws; ws += (size_t)NB * ND * 2;
  unsigned short* wt[8];
  for (int i = 0; i < 8; ++i) { wt[i] = (unsigned short*)ws; ws += (size_t)ND * NH * 2; }
  unsigned short *s1W1t = wt[0], *t1W1t = wt[1], *s1W2t = wt[2], *t1W2t = wt[3];
  unsigned short *s2W1t = wt[4], *t2W1t = wt[5], *s2W2t = wt[6], *t2W2t = wt[7];

  float* y_out   = (float*)d_out;                    // NB x 2048
  float* log_det = y_out + (size_t)NB * (2 * ND);    // NB

  const dim3 blk(256);
  const dim3 gW1(NH / 32, ND / 32);   // W1: K=1024, N=4096 -> (128, 32)
  const dim3 gW2(ND / 32, NH / 32);   // W2: K=4096, N=1024 -> (32, 128)
  const dim3 g1(NH / BN, NB / BM);    // layer-1 GEMMs: 32 x 64 blocks
  const dim3 g2(ND / BN, NB / BM);    // layer-2 GEMMs:  8 x 64 blocks

  // Weight prep (fp32 [KxN] -> bf16 [NxK]) + x2 conversion.
  transpose_cvt_kernel<<<gW1, blk, 0, stream>>>(s1_W1, s1W1t, ND, NH);
  transpose_cvt_kernel<<<gW1, blk, 0, stream>>>(t1_W1, t1W1t, ND, NH);
  transpose_cvt_kernel<<<gW2, blk, 0, stream>>>(s1_W2, s1W2t, NH, ND);
  transpose_cvt_kernel<<<gW2, blk, 0, stream>>>(t1_W2, t1W2t, NH, ND);
  transpose_cvt_kernel<<<gW1, blk, 0, stream>>>(s2_W1, s2W1t, ND, NH);
  transpose_cvt_kernel<<<gW1, blk, 0, stream>>>(t2_W1, t2W1t, ND, NH);
  transpose_cvt_kernel<<<gW2, blk, 0, stream>>>(s2_W2, s2W2t, NH, ND);
  transpose_cvt_kernel<<<gW2, blk, 0, stream>>>(t2_W2, t2W2t, NH, ND);
  cvt_x2_kernel<<<(NB * ND) / 256, blk, 0, stream>>>(x, x2_bf);

  // Phase 1: s2v = tanh(mlp(x2; s1)), t2v = mlp(x2; t1)
  gemm_bf16_wmma<true,  true ><<<g1, blk, 0, stream>>>(x2_bf, s1W1t, s1_b1, h_s,   NH, ND);
  gemm_bf16_wmma<true,  true ><<<g1, blk, 0, stream>>>(x2_bf, t1W1t, t1_b1, h_t,   NH, ND);
  gemm_bf16_wmma<false, false><<<g2, blk, 0, stream>>>(h_s,   s1W2t, s1_b2, s_pre, ND, NH);
  gemm_bf16_wmma<false, false><<<g2, blk, 0, stream>>>(h_t,   t1W2t, t1_b2, t_val, ND, NH);

  // y1 = x1 * exp(s2v) + t2v ; log_det = sum(s2v)
  coupling_kernel<<<NB, blk, 0, stream>>>(x, s_pre, t_val, y_out, y1_bf, log_det, 0, 0);

  // Phase 2: s1v = tanh(mlp(y1; s2)), t1v = mlp(y1; t2)
  gemm_bf16_wmma<true,  true ><<<g1, blk, 0, stream>>>(y1_bf, s2W1t, s2_b1, h_s,   NH, ND);
  gemm_bf16_wmma<true,  true ><<<g1, blk, 0, stream>>>(y1_bf, t2W1t, t2_b1, h_t,   NH, ND);
  gemm_bf16_wmma<false, false><<<g2, blk, 0, stream>>>(h_s,   s2W2t, s2_b2, s_pre, ND, NH);
  gemm_bf16_wmma<false, false><<<g2, blk, 0, stream>>>(h_t,   t2W2t, t2_b2, t_val, ND, NH);

  // y2 = x2 * exp(s1v) + t1v ; log_det += sum(s1v)
  coupling_kernel<<<NB, blk, 0, stream>>>(x, s_pre, t_val, y_out, nullptr, log_det, ND, 1);
}
